// ModulatedConv1d_30296699306039
// MI455X (gfx1250) — matrix operand order repaired and checked by reading, not compiled, not run on hardware
//
#include <hip/hip_runtime.h>

typedef __attribute__((ext_vector_type(16))) _Float16 v16h;
typedef __attribute__((ext_vector_type(8)))  _Float16 v8h;
typedef __attribute__((ext_vector_type(8)))  float    v8f;

#define B_  16
#define I_  128
#define O_  128
#define T_  256
#define L_  16384
#define EPS_ 1e-8f

// ---------------------------------------------------------------------------
// Kernel 1: scale[b][i] = t[b,:] . mod_w[i,:] + mod_b[i] + 1
// ---------------------------------------------------------------------------
__global__ void scale_kernel(const float* __restrict__ t,
                             const float* __restrict__ mod_w,
                             const float* __restrict__ mod_b,
                             float* __restrict__ scale) {
  int idx = blockIdx.x * blockDim.x + threadIdx.x;
  if (idx >= B_ * I_) return;
  int b = idx / I_;
  int i = idx - b * I_;
  const float* trow = t + b * T_;
  const float* wrow = mod_w + i * T_;
  float acc = 0.f;
#pragma unroll 8
  for (int k = 0; k < T_; ++k) acc += trow[k] * wrow[k];
  scale[idx] = acc + mod_b[i] + 1.0f;
}

// ---------------------------------------------------------------------------
// Kernel 2: Af16[b][o][i] = f16( conv_w[o,i]*scale[b,i] * rsqrt(sum_i(.)^2+eps) )
// one block per (o, b); 128 threads (one per i)
// ---------------------------------------------------------------------------
__global__ void modw_kernel(const float* __restrict__ conv_w,
                            const float* __restrict__ scale,
                            _Float16* __restrict__ Af) {
  const int o = blockIdx.x;
  const int b = blockIdx.y;
  const int i = threadIdx.x;

  float v = conv_w[o * I_ + i] * scale[b * I_ + i];

  __shared__ float red[I_];
  red[i] = v * v;
  __syncthreads();
#pragma unroll
  for (int s = I_ / 2; s > 0; s >>= 1) {
    if (i < s) red[i] += red[i + s];
    __syncthreads();
  }
  float demod = rsqrtf(red[0] + EPS_);
  Af[((size_t)(b * O_ + o)) * I_ + i] = (_Float16)(v * demod);
}

// ---------------------------------------------------------------------------
// Kernel 3: out[b] (128 x 16384) = Af16[b] (128x128) @ f16(x[b]) (128x16384)
// 256 threads = 8 waves; each wave: 16-column strip, all 8 M-tiles.
// A staged in LDS (32 KB f16), B converted f32->f16 in registers.
// ---------------------------------------------------------------------------
__global__ void __launch_bounds__(256)
gemm_kernel(const _Float16* __restrict__ Af,
            const float* __restrict__ x,
            float* __restrict__ out) {
  __shared__ __align__(16) _Float16 Ash[O_ * I_];  // 32 KB

  const int b   = blockIdx.y;
  const int tid = threadIdx.x;

  // cooperative stage of A tile: 32768 B = 2048 uint4, 8 per thread
  {
    const uint4* src = (const uint4*)(Af + (size_t)b * O_ * I_);
    uint4* dst = (uint4*)Ash;
#pragma unroll
    for (int r = 0; r < (O_ * I_ * (int)sizeof(_Float16) / 16) / 256; ++r)
      dst[r * 256 + tid] = src[r * 256 + tid];
  }
  __syncthreads();

  const int wave = tid >> 5;
  const int lane = tid & 31;
  const int h    = lane >> 4;   // half-wave selector
  const int nlo  = lane & 15;
  const int col  = blockIdx.x * 128 + wave * 16 + nlo;

  const float* xb = x + (size_t)b * I_ * L_;

  v8f acc[8];
#pragma unroll
  for (int m = 0; m < 8; ++m) acc[m] = {};

#pragma unroll
  for (int ks = 0; ks < 4; ++ks) {
    const int kbase = ks * 32;

    // prefetch next K-step of x (global_prefetch_b8)
    if (ks < 3)
      __builtin_prefetch(&xb[(size_t)(kbase + 32 + 16 * h) * L_ + col], 0, 0);

    // B fragment: column `col`; lanes 0-15 hold K=kbase+{0..7,16..23}, lanes 16-31 K=kbase+{8..15,24..31}
    v16h bf;
#pragma unroll
    for (int q = 0; q < 16; ++q) {
      const int kk = kbase + 8 * h + (q & 7) + ((q >> 3) << 4);
      bf[q] = (_Float16)xb[(size_t)kk * L_ + col];
    }

#pragma unroll
    for (int m = 0; m < 8; ++m) {
      const int r = m * 16 + nlo;  // A row for this lane
      // 16-bit A 16x32 layout: lanes 0-15 K = {0..7, 16..23}, lanes 16-31 K = {8..15, 24..31}
      const v8h a_lo = *(const v8h*)(&Ash[r * I_ + kbase + 8 * h]);
      const v8h a_hi = *(const v8h*)(&Ash[r * I_ + kbase + 16 + 8 * h]);
      const v16h af = __builtin_shufflevector(a_lo, a_hi,
          0, 1, 2, 3, 4, 5, 6, 7, 8, 9, 10, 11, 12, 13, 14, 15);
      acc[m] = __builtin_amdgcn_wmma_f32_16x16x32_f16(
          /*neg_a=*/false, af, /*neg_b=*/false, bf,
          /*c_mod=*/(short)0, acc[m], /*reuse_a=*/false, /*reuse_b=*/false);
    }
  }

  // C/D f32 layout: VGPR v -> row (m*16 + 8*h + v), col = blockcol + nlo
  float* ob = out + (size_t)b * O_ * L_;
#pragma unroll
  for (int m = 0; m < 8; ++m) {
    const int rbase = m * 16 + 8 * h;
#pragma unroll
    for (int v = 0; v < 8; ++v) {
      ob[(size_t)(rbase + v) * L_ + col] = acc[m][v];
    }
  }
}

// ---------------------------------------------------------------------------
extern "C" void kernel_launch(void* const* d_in, const int* in_sizes, int n_in,
                              void* d_out, int out_size, void* d_ws, size_t ws_size,
                              hipStream_t stream) {
  const float* x      = (const float*)d_in[0];
  const float* t      = (const float*)d_in[1];
  const float* conv_w = (const float*)d_in[2];
  const float* mod_w  = (const float*)d_in[3];
  const float* mod_b  = (const float*)d_in[4];
  float* out = (float*)d_out;

  float*    scale = (float*)d_ws;
  _Float16* Af    = (_Float16*)((char*)d_ws + (size_t)B_ * I_ * sizeof(float));

  scale_kernel<<<(B_ * I_ + 255) / 256, 256, 0, stream>>>(t, mod_w, mod_b, scale);
  modw_kernel<<<dim3(O_, B_), 128, 0, stream>>>(conv_w, scale, Af);
  gemm_kernel<<<dim3(L_ / 128, B_), 256, 0, stream>>>(Af, x, out);

  (void)in_sizes; (void)n_in; (void)out_size; (void)ws_size;
}
